// GQAttention_LR_33904471835494
// MI455X (gfx1250) — matrix operand order repaired and checked by reading, not compiled, not run on hardware
//
#include <hip/hip_runtime.h>
#include <hip/hip_bf16.h>

typedef __attribute__((ext_vector_type(16))) __bf16 bf16x16;
typedef __attribute__((ext_vector_type(8)))  __bf16 bf16x8;
typedef __attribute__((ext_vector_type(8)))  float  f32x8;

typedef int i32x4 __attribute__((vector_size(16)));
typedef __attribute__((address_space(1))) i32x4 as1_i32x4;   // global
typedef __attribute__((address_space(3))) i32x4 as3_i32x4;   // LDS

#define NB    2
#define SEQ   2048
#define DK    2048
#define NHQ   32
#define NHKV  8
#define HDIM  64

#define LDA_PAD 40   // 32 + 8 bf16: 80B row stride = 20 banks, conflict-free for 16-row strides

__device__ __forceinline__ __bf16 f2bf(float f) {
  unsigned u = __builtin_bit_cast(unsigned, f);
  unsigned r = u + 0x7FFFu + ((u >> 16) & 1u);   // round-to-nearest-even
  unsigned short s = (unsigned short)(r >> 16);
  return __builtin_bit_cast(__bf16, s);
}

__device__ __forceinline__ bf16x16 cat8(bf16x8 lo, bf16x8 hi) {
  return __builtin_shufflevector(lo, hi, 0,1,2,3,4,5,6,7,8,9,10,11,12,13,14,15);
}

__device__ __forceinline__ f32x8 wmma_bf16(bf16x16 a, bf16x16 b, f32x8 c) {
  return __builtin_amdgcn_wmma_f32_16x16x32_bf16(false, a, false, b, (short)0, c, false, false);
}

// 16-byte global -> LDS copy, async (ASYNCcnt) via the gfx1250 instruction when available.
__device__ __forceinline__ void cp16_g2l(const __bf16* g, __bf16* l) {
#if __has_builtin(__builtin_amdgcn_global_load_async_to_lds_b128)
  __builtin_amdgcn_global_load_async_to_lds_b128((as1_i32x4*)g, (as3_i32x4*)l, 0, 0);
#else
  *(bf16x8*)l = *(const bf16x8*)g;
#endif
}

__device__ __forceinline__ void wait_async() {
#if __has_builtin(__builtin_amdgcn_s_wait_asynccnt)
  __builtin_amdgcn_s_wait_asynccnt(0);
#else
  asm volatile("s_wait_asynccnt 0x0" ::: "memory");
#endif
}

// ---------------------------------------------------------------- f32 -> bf16
__global__ void cvt_bf16(const float* __restrict__ in, __bf16* __restrict__ out, long n) {
  long i = (long)blockIdx.x * blockDim.x + threadIdx.x;
  long stride = (long)gridDim.x * blockDim.x;
  for (; i < n; i += stride) out[i] = f2bf(in[i]);
}

// ---------------------------------------------------------------- WMMA GEMM
// C(M x N) = A(M x K) * B^T, B row-major (N x K).
// Block = 256 threads = 8 waves as 4(M) x 2(N); block tile 64M x 128N staged via LDS.
// MODE 0: f32 row-major out. MODE 1: RoPE -> bf16 [b][h][s][hd]. MODE 2: bf16 [b][h][hd][s].
template<int MODE>
__global__ __launch_bounds__(256) void gemm_wmma(
    const __bf16* __restrict__ A, const __bf16* __restrict__ Bw,
    float* __restrict__ outF, __bf16* __restrict__ outB,
    int M, int N, int Kd, int nheads) {
  __shared__ alignas(16) __bf16 At[64 * LDA_PAD];    // 5  KB
  __shared__ alignas(16) __bf16 Bt[128 * LDA_PAD];   // 10 KB

  const int tid  = threadIdx.x;
  const int lane = tid & 31;
  const int col  = lane & 15;
  const int half = lane >> 4;
  const int w    = tid >> 5;
  const int wm   = w >> 1;            // 0..3
  const int wn   = w & 1;             // 0..1
  const int nblk = N >> 7;
  const int bm   = blockIdx.x / nblk;
  const int bn   = blockIdx.x % nblk;

  // staging assignment: 16B chunks; A: 64 rows x 4 chunks (1/thread); B: 128 rows (2/thread)
  const int srow = tid >> 2, schk = (tid & 3) * 8;
  const __bf16* ga  = A  + (size_t)(bm * 64 + srow) * Kd + schk;
  const __bf16* gb0 = Bw + (size_t)(bn * 128 + srow) * Kd + schk;
  const __bf16* gb1 = Bw + (size_t)(bn * 128 + 64 + srow) * Kd + schk;
  __bf16* la  = &At[srow * LDA_PAD + schk];
  __bf16* lb0 = &Bt[srow * LDA_PAD + schk];
  __bf16* lb1 = &Bt[(64 + srow) * LDA_PAD + schk];

  // fragment read addresses (contiguous k = 16*half + s for both A and B)
  const __bf16* fa = &At[(wm * 16 + col) * LDA_PAD + (half ? 16 : 0)];
  const __bf16* fb[4];
#pragma unroll
  for (int nt = 0; nt < 4; ++nt)
    fb[nt] = &Bt[(wn * 64 + nt * 16 + col) * LDA_PAD + (half ? 16 : 0)];

  f32x8 acc[4] = {};
  for (int k = 0; k < Kd; k += 32) {
    cp16_g2l(ga + k, la);
    cp16_g2l(gb0 + k, lb0);
    cp16_g2l(gb1 + k, lb1);
    wait_async();
    __syncthreads();
    bf16x16 a = cat8(*(const bf16x8*)fa, *(const bf16x8*)(fa + 8));
#pragma unroll
    for (int nt = 0; nt < 4; ++nt) {
      bf16x16 b = *(const bf16x16*)fb[nt];
      acc[nt] = wmma_bf16(a, b, acc[nt]);
    }
    __syncthreads();
  }

  if (MODE == 0) {
#pragma unroll
    for (int nt = 0; nt < 4; ++nt) {
      int cg = bn * 128 + wn * 64 + nt * 16 + col;
#pragma unroll
      for (int r = 0; r < 8; ++r) {
        int m = bm * 64 + wm * 16 + r + 8 * half;
        outF[(size_t)m * N + cg] = acc[nt][r];
      }
    }
  } else if (MODE == 1) {
#pragma unroll
    for (int nt = 0; nt < 4; ++nt) {
      int   cg = bn * 128 + wn * 64 + nt * 16 + col;
      int   h  = cg >> 6, hd = cg & 63;
      float inv = __powf(10000.0f, -(float)(hd >> 1) * (1.0f / 32.0f));
#pragma unroll
      for (int r = 0; r < 8; ++r) {
        int   m = bm * 64 + wm * 16 + r + 8 * half;
        int   b = m >> 11, s = m & (SEQ - 1);
        float v = acc[nt][r];
        float partner = __shfl_xor(v, 1, 32);   // other half of the RoPE pair
        float ang = (float)s * inv, sn, cs;
        __sincosf(ang, &sn, &cs);
        float o = (hd & 1) ? (v * cs + partner * sn) : (v * cs - partner * sn);
        outB[(((size_t)(b * nheads + h) * SEQ + s) << 6) + hd] = f2bf(o);
      }
    }
  } else {
#pragma unroll
    for (int nt = 0; nt < 4; ++nt) {
      int cg = bn * 128 + wn * 64 + nt * 16 + col;
      int h  = cg >> 6, hd = cg & 63;
#pragma unroll
      for (int r = 0; r < 8; ++r) {
        int m = bm * 64 + wm * 16 + r + 8 * half;
        int b = m >> 11, s = m & (SEQ - 1);
        outB[((size_t)(b * nheads + h) * HDIM + hd) * SEQ + s] = f2bf(acc[nt][r]);
      }
    }
  }
}

// ---------------------------------------------------------------- flash attention
// One wave = one 16-query tile of one head; causal key tiles processed in pairs.
__global__ void flash_attn(const __bf16* __restrict__ qb, const __bf16* __restrict__ kb,
                           const __bf16* __restrict__ vt, const int* __restrict__ am,
                           __bf16* __restrict__ zb) {
  __shared__ alignas(16) __bf16 Pb[8][16 * 32];
  const int lane = threadIdx.x & 31;
  const int col  = lane & 15;
  const int half = lane >> 4;
  const int w    = threadIdx.x >> 5;
  const int wid  = (blockIdx.x * blockDim.x + threadIdx.x) >> 5;
  const int qt = wid & 127;
  const int h  = (wid >> 7) & 31;
  const int b  = wid >> 12;
  const int hk = h >> 2;

  const __bf16* qrow = qb + ((((size_t)(b * NHQ + h) * SEQ) + qt * 16 + col) << 6) + (half ? 16 : 0);
  bf16x16 aq[2];
#pragma unroll
  for (int ks = 0; ks < 2; ++ks)
    aq[ks] = cat8(*(const bf16x8*)(qrow + ks * 32),
                  *(const bf16x8*)(qrow + ks * 32 + 8));

  const __bf16* kh  = kb + (((size_t)(b * NHKV + hk) * SEQ) << 6);
  const __bf16* vh  = vt + ((size_t)(b * NHKV + hk) * HDIM) * SEQ;
  const int*    amb = am + b * SEQ;

  f32x8 acc[4] = {};
  float mrow[8], lrow[8];
#pragma unroll
  for (int r = 0; r < 8; ++r) { mrow[r] = -1e30f; lrow[r] = 0.0f; }

  const int rowbase = qt * 16 + 8 * half;

  for (int kt = 0; kt <= qt; kt += 2) {
    const bool has2 = (kt + 1) <= qt;
    f32x8 s0 = {}, s1 = {};
    {
      int key = kt * 16 + col;
      const __bf16* kr = kh + ((size_t)key << 6) + (half ? 16 : 0);
#pragma unroll
      for (int ks = 0; ks < 2; ++ks)
        s0 = wmma_bf16(aq[ks], *(const bf16x16*)(kr + ks * 32), s0);
      bool pad = amb[key] != 0;
#pragma unroll
      for (int r = 0; r < 8; ++r)
        s0[r] = (pad && key <= rowbase + r) ? s0[r] * 0.125f : -1e30f;
    }
    if (has2) {
      int key = (kt + 1) * 16 + col;
      const __bf16* kr = kh + ((size_t)key << 6) + (half ? 16 : 0);
#pragma unroll
      for (int ks = 0; ks < 2; ++ks)
        s1 = wmma_bf16(aq[ks], *(const bf16x16*)(kr + ks * 32), s1);
      bool pad = amb[key] != 0;
#pragma unroll
      for (int r = 0; r < 8; ++r)
        s1[r] = (pad && key <= rowbase + r) ? s1[r] * 0.125f : -1e30f;
    } else {
#pragma unroll
      for (int r = 0; r < 8; ++r) s1[r] = -1e30f;
    }

    float alpha[8];
#pragma unroll
    for (int r = 0; r < 8; ++r) {
      float mx = fmaxf(s0[r], s1[r]);
#pragma unroll
      for (int o = 1; o < 16; o <<= 1) mx = fmaxf(mx, __shfl_xor(mx, o, 32));
      float mnew = fmaxf(mrow[r], mx);
      alpha[r] = __expf(mrow[r] - mnew);
      float p0 = __expf(s0[r] - mnew);
      float p1 = __expf(s1[r] - mnew);
      float rs = p0 + p1;
#pragma unroll
      for (int o = 1; o < 16; o <<= 1) rs += __shfl_xor(rs, o, 32);
      lrow[r] = lrow[r] * alpha[r] + rs;
      mrow[r] = mnew;
      Pb[w][(r + 8 * half) * 32 + col]      = f2bf(p0);
      Pb[w][(r + 8 * half) * 32 + 16 + col] = f2bf(p1);
    }
#pragma unroll
    for (int nt = 0; nt < 4; ++nt)
#pragma unroll
      for (int r = 0; r < 8; ++r)
        acc[nt][r] *= alpha[r];

    // LDS relayout: C-layout P -> A-fragment layout for the PV matmul
    asm volatile("s_wait_dscnt 0x0" ::: "memory");
    bf16x16 pA = cat8(*(const bf16x8*)&Pb[w][col * 32 + (half ? 16 : 0)],
                      *(const bf16x8*)&Pb[w][col * 32 + (half ? 16 : 0) + 8]);
    int kt16 = kt + half;            // lanes 0-15: keys of tile kt; 16-31: tile kt+1
    if (kt16 > qt) kt16 = qt;        // fully-masked half: P==0, clamp to stay in-bounds
#pragma unroll
    for (int nt = 0; nt < 4; ++nt) {
      const __bf16* vr = vh + (size_t)(nt * 16 + col) * SEQ + kt16 * 16;
      acc[nt] = wmma_bf16(pA, *(const bf16x16*)vr, acc[nt]);
    }
  }

#pragma unroll
  for (int r = 0; r < 8; ++r) {
    float invl = 1.0f / lrow[r];
    int s = qt * 16 + r + 8 * half;
    __bf16* zr = zb + ((size_t)(b * SEQ + s) << 11) + h * HDIM;
#pragma unroll
    for (int nt = 0; nt < 4; ++nt)
      zr[nt * 16 + col] = f2bf(acc[nt][r] * invl);
  }
}

// ---------------------------------------------------------------- launch
extern "C" void kernel_launch(void* const* d_in, const int* in_sizes, int n_in,
                              void* d_out, int out_size, void* d_ws, size_t ws_size,
                              hipStream_t stream) {
  const float* x  = (const float*)d_in[0];
  const int*   am = (const int*)d_in[1];
  const float* Wq = (const float*)d_in[2];
  const float* Wk = (const float*)d_in[3];
  const float* Wv = (const float*)d_in[4];
  const float* Wo = (const float*)d_in[5];
  float* out = (float*)d_out;

  char* p = (char*)d_ws;
  auto carve = [&](size_t nelem) {
    __bf16* q = (__bf16*)p;
    p += ((nelem * 2 + 255) & ~(size_t)255);
    return q;
  };
  const size_t NX = (size_t)NB * SEQ * DK;             // 8,388,608
  __bf16* xb  = carve(NX);                             // x  bf16
  __bf16* wqb = carve((size_t)DK * DK);                // Wq bf16
  __bf16* wkb = carve((size_t)NHKV * HDIM * DK);       // Wk bf16
  __bf16* wvb = carve((size_t)NHKV * HDIM * DK);       // Wv bf16
  __bf16* wob = carve((size_t)DK * DK);                // Wo bf16
  __bf16* qb  = carve(NX);                             // Q roped, [b][h][s][hd]
  __bf16* kbf = carve((size_t)NB * NHKV * SEQ * HDIM); // K roped, [b][hk][s][hd]
  __bf16* vtb = carve((size_t)NB * NHKV * HDIM * SEQ); // V^T,     [b][hk][hd][s]
  __bf16* zbb = carve(NX);                             // attn out [b][s][h*hd]

  cvt_bf16<<<4096, 256, 0, stream>>>(x,  xb,  (long)NX);
  cvt_bf16<<<4096, 256, 0, stream>>>(Wq, wqb, (long)DK * DK);
  cvt_bf16<<<2048, 256, 0, stream>>>(Wk, wkb, (long)NHKV * HDIM * DK);
  cvt_bf16<<<2048, 256, 0, stream>>>(Wv, wvb, (long)NHKV * HDIM * DK);
  cvt_bf16<<<4096, 256, 0, stream>>>(Wo, wob, (long)DK * DK);

  const int M = NB * SEQ;  // 4096 rows
  // Q projection + RoPE : blocks = (M/64) * (N/128)
  gemm_wmma<1><<<(M / 64) * (2048 / 128), 256, 0, stream>>>(xb, wqb, nullptr, qb,  M, 2048, DK, NHQ);
  // K projection + RoPE
  gemm_wmma<1><<<(M / 64) * (512 / 128),  256, 0, stream>>>(xb, wkb, nullptr, kbf, M, 512,  DK, NHKV);
  // V projection, transposed store
  gemm_wmma<2><<<(M / 64) * (512 / 128),  256, 0, stream>>>(xb, wvb, nullptr, vtb, M, 512,  DK, NHKV);
  // causal GQA flash attention
  flash_attn<<<(NB * NHQ * (SEQ / 16)) / 8, 256, 0, stream>>>(qb, kbf, vtb, am, zbb);
  // output projection -> f32
  gemm_wmma<0><<<(M / 64) * (2048 / 128), 256, 0, stream>>>(zbb, wob, out, nullptr, M, 2048, DK, 0);
}
